// C_Net_25829933318366
// MI455X (gfx1250) — matrix operand order repaired and checked by reading, not compiled, not run on hardware
//
#include <hip/hip_runtime.h>
#include <hip/hip_bf16.h>

// Problem constants (fixed by setup_inputs: B=2, C=256, H=W=64, 12 classes).
constexpr int B_    = 2;
constexpr int NCLS  = 12;     // class channels, loop uses 1..11
constexpr int C_    = 256;    // feature channels
constexpr int S_    = 4096;   // 64*64 pixels
constexpr float MASK_OFF = 1e12f;  // folded softmax mask offset
constexpr float SHIFT    = 2.0f;   // fixed softmax shift (corr is a cosine, |corr|<=~1)

typedef __attribute__((ext_vector_type(16))) __bf16 v16bf;
typedef __attribute__((ext_vector_type(8)))  __bf16 v8bf;
typedef __attribute__((ext_vector_type(8)))  float  v8f;

// ---------- small device helpers ----------
__device__ inline unsigned short f2bf(float f) {
    // round-to-nearest-even float -> bf16 (bit pattern)
    unsigned int u = __float_as_uint(f);
    unsigned int r = (u + 0x7FFFu + ((u >> 16) & 1u)) >> 16;
    return (unsigned short)r;
}

__device__ inline float wavesum32(float v) {
    v += __shfl_xor(v, 16, 32);
    v += __shfl_xor(v, 8, 32);
    v += __shfl_xor(v, 4, 32);
    v += __shfl_xor(v, 2, 32);
    v += __shfl_xor(v, 1, 32);
    return v;
}
__device__ inline float grpsum16(float v) {   // reduce over 16-lane half-wave group
    v += __shfl_xor(v, 8, 32);
    v += __shfl_xor(v, 4, 32);
    v += __shfl_xor(v, 2, 32);
    v += __shfl_xor(v, 1, 32);
    return v;
}
__device__ inline float block_reduce_sum(float v, float* red) { // blockDim.x == 256
    int tid = threadIdx.x;
    red[tid] = v; __syncthreads();
    for (int o = 128; o >= 1; o >>= 1) { if (tid < o) red[tid] += red[tid + o]; __syncthreads(); }
    float r = red[0]; __syncthreads();
    return r;
}

// ---------- kernel 1: transpose/convert features to bf16 [B,S,C], per-pixel norm^2 ----------
__global__ void prep_kernel(const float* __restrict__ cf, const float* __restrict__ rf,
                            float* __restrict__ normC2, float* __restrict__ normL2,
                            unsigned short* __restrict__ qT, unsigned short* __restrict__ kT) {
    int idx = blockIdx.x * 256 + threadIdx.x;           // b*S + p
    int b = idx / S_, p = idx % S_;
    size_t fbase = (size_t)b * C_ * S_;
    size_t tbase = (size_t)idx * C_;
    float nc = 0.f, nl = 0.f;
    for (int ch = 0; ch < C_; ch++) {
        float a = cf[fbase + (size_t)ch * S_ + p];
        float r = rf[fbase + (size_t)ch * S_ + p];
        nc += a * a; nl += r * r;
        qT[tbase + ch] = f2bf(a);
        kT[tbase + ch] = f2bf(r);
    }
    normC2[idx] = nc;
    normL2[idx] = nl;
}

// ---------- kernel 2: per (b,class) masked means + scalar stats ----------
__global__ void stats_kernel(const float* __restrict__ cf, const float* __restrict__ rf,
                             const float* __restrict__ cmask, const float* __restrict__ lmask,
                             float* __restrict__ muC, float* __restrict__ muL,
                             float* __restrict__ stats) {
    int cls = blockIdx.x + 1;
    int b = blockIdx.y;
    __shared__ float cmL[S_];
    __shared__ float lmL[S_];
    __shared__ float red[256];
    __shared__ float muCL[C_], muLL[C_];
    int tid = threadIdx.x;
    size_t mbase = ((size_t)b * NCLS + cls) * S_;

    float c0 = 0.f, c1 = 0.f;
    for (int i = tid; i < S_; i += 256) {
        float m1 = cmask[mbase + i]; cmL[i] = m1; c0 += m1;
        float m2 = lmask[mbase + i]; lmL[i] = m2; c1 += m2;
    }
    __syncthreads();
    float nC = block_reduce_sum(c0, red);
    float nL = block_reduce_sum(c1, red);
    float dC = fmaxf(nC, 1.f), dL = fmaxf(nL, 1.f);

    int wave = tid >> 5, lane = tid & 31;
    size_t cbase = ((size_t)b * NCLS + cls) * C_;
    size_t fbase = (size_t)b * C_ * S_;
    for (int ch = wave; ch < C_; ch += 8) {
        float sC = 0.f, sL = 0.f;
        for (int p = lane; p < S_; p += 32) {
            float fa = cf[fbase + (size_t)ch * S_ + p];
            float fr = rf[fbase + (size_t)ch * S_ + p];
            sC += fa * cmL[p];
            sL += fr * lmL[p];
        }
        sC = wavesum32(sC); sL = wavesum32(sL);
        if (lane == 0) {
            float m1 = sC / dC, m2 = sL / dL;
            muCL[ch] = m1; muLL[ch] = m2;
            muC[cbase + ch] = m1; muL[cbase + ch] = m2;
        }
    }
    __syncthreads();
    float a = muCL[tid], bb = muLL[tid];      // 256 threads == C_
    float cc  = block_reduce_sum(a * bb, red);
    float mc2 = block_reduce_sum(a * a, red);
    float ml2 = block_reduce_sum(bb * bb, red);
    if (tid == 0) {
        size_t sb = ((size_t)b * NCLS + cls) * 8;
        stats[sb + 0] = nC;
        stats[sb + 1] = nL;
        stats[sb + 2] = cc;
        stats[sb + 3] = mc2;
        stats[sb + 4] = ml2;
        stats[sb + 5] = (nC > 1.f && nL > 1.f) ? 1.f : 0.f;
    }
}

// ---------- kernel 3: per (b,class,pixel) correction scalars ----------
// Row side: betaP (cf_p . muL), alphaP (guarded inverse centered norm).
// Key side: packed kq = { betaQ + maskOffset, alphaQ } so the attention inner
// loop needs a single b64 load and zero branches (masked keys underflow to 0).
__global__ void pixstats_kernel(const float* __restrict__ cf, const float* __restrict__ rf,
                                const float* __restrict__ muC, const float* __restrict__ muL,
                                const float* __restrict__ stats, const float* __restrict__ lmask,
                                const float* __restrict__ normC2, const float* __restrict__ normL2,
                                float* __restrict__ betaP, float* __restrict__ alphaP,
                                float2* __restrict__ kq) {
    int cls = blockIdx.y + 1;
    int b = blockIdx.z;
    int p = blockIdx.x * 256 + threadIdx.x;
    __shared__ float mC[C_], mL[C_];
    size_t cbase = ((size_t)b * NCLS + cls) * C_;
    mC[threadIdx.x] = muC[cbase + threadIdx.x];
    mL[threadIdx.x] = muL[cbase + threadIdx.x];
    __syncthreads();
    size_t fbase = (size_t)b * C_ * S_;
    float dCC = 0.f, dCL = 0.f, dQC = 0.f, dQL = 0.f;
    for (int ch = 0; ch < C_; ch++) {
        float a = cf[fbase + (size_t)ch * S_ + p];
        float r = rf[fbase + (size_t)ch * S_ + p];
        dCC += a * mC[ch]; dCL += a * mL[ch];
        dQC += r * mC[ch]; dQL += r * mL[ch];
    }
    size_t sb = ((size_t)b * NCLS + cls) * 8;
    float mc2 = stats[sb + 3], ml2 = stats[sb + 4];
    float np2 = normC2[b * S_ + p] - 2.f * dCC + mc2;
    float nq2 = normL2[b * S_ + p] - 2.f * dQL + ml2;
    float npn = sqrtf(fmaxf(np2, 0.f));
    float nqn = sqrtf(fmaxf(nq2, 0.f));
    size_t ob = ((size_t)b * NCLS + cls) * S_ + p;
    betaP[ob]  = dCL;
    alphaP[ob] = (npn > 0.f) ? 1.f / npn : 1.f;
    bool lmv = lmask[ob] > 0.f;            // lmask shares [B,NCLS,S] indexing
    kq[ob] = make_float2(dQC + (lmv ? 0.f : MASK_OFF),
                         (nqn > 0.f) ? 1.f / nqn : 1.f);
}

// ---------- kernel 4: owner map + canvas init + packed image ----------
__global__ void owner_kernel(const float* __restrict__ cmask, const float* __restrict__ stats,
                             const float* __restrict__ img,
                             int* __restrict__ owner, float4* __restrict__ imgPk,
                             float* __restrict__ out) {
    int idx = blockIdx.x * 256 + threadIdx.x;            // b*S + p
    int b = idx / S_, p = idx % S_;
    int own = 0;
    for (int cls = 1; cls < NCLS; cls++) {
        float v = stats[((size_t)b * NCLS + cls) * 8 + 5];
        if (v != 0.f && cmask[((size_t)b * NCLS + cls) * S_ + p] > 0.f) own = cls;
    }
    owner[idx] = own;
    imgPk[idx] = make_float4(img[((size_t)b * 3 + 0) * S_ + p],
                             img[((size_t)b * 3 + 1) * S_ + p],
                             img[((size_t)b * 3 + 2) * S_ + p], 0.f);
    out[((size_t)b * 3 + 0) * S_ + p] = -1.f;
    out[((size_t)b * 3 + 1) * S_ + p] = -1.f;
    out[((size_t)b * 3 + 2) * S_ + p] = -1.f;
}

// ---------- kernel 5: fused WMMA attention colorization (fixed-shift softmax) ----------
// Grid: (S/16, 11, B). 128 threads = 4 waves; each wave strides 16-key tiles.
// corr is a cosine similarity => bounded, so exp(corr - 2) is a branch-free,
// shuffle-free, exact softmax weight; masked keys underflow to exactly 0.
__global__ void __launch_bounds__(128)
attn_kernel(const unsigned short* __restrict__ qT, const unsigned short* __restrict__ kT,
            const float* __restrict__ betaP, const float* __restrict__ alphaP,
            const float2* __restrict__ kq, const float4* __restrict__ imgPk,
            const float* __restrict__ stats, const int* __restrict__ owner,
            float* __restrict__ out) {
    const int cls = blockIdx.y + 1;
    const int b   = blockIdx.z;
    const int p0  = blockIdx.x * 16;
    const int tid = threadIdx.x;
    const int lane = tid & 31;
    const int wave = tid >> 5;
    const int bS = b * S_;

    size_t sb = ((size_t)b * NCLS + cls) * 8;
    if (stats[sb + 5] == 0.f) return;                    // class invalid: uniform exit
    const float cc = stats[sb + 2];

    __shared__ float sBP[16], sAP[16];
    __shared__ float mS[4][16], mCol[4][16][3];
    __shared__ int sFlag;

    const size_t pcBase = ((size_t)b * NCLS + cls) * S_;
    if (tid == 0) sFlag = 0;
    __syncthreads();
    if (tid < 16) {
        sBP[tid] = betaP[pcBase + p0 + tid];
        sAP[tid] = alphaP[pcBase + p0 + tid];
        if (owner[bS + p0 + tid] == cls) sFlag = 1;
    }
    __syncthreads();
    if (!sFlag) return;                                  // no pixel here owned by this class

    // per-lane row constants (C/D layout: VGPR j holds row j (lanes 0-15), row j+8 (lanes 16-31))
    const int hh = lane >> 4;                            // half-wave select
    float cj[8], raP[8];
#pragma unroll
    for (int j = 0; j < 8; j++) {
        int r = j + hh * 8;
        cj[j]  = cc - sBP[r];                            // folded (cc - betaP[row])
        raP[j] = sAP[r];
    }

    // A fragments: 16 query rows x 256 channels, 8 chunks of K=32 (16-bit A layout)
    const int arow = lane & 15;
    const size_t aBase = ((size_t)bS + p0 + arow) * C_;
    const int aOff = (lane < 16) ? 0 : 8;
    v16bf afr[8];
#pragma unroll
    for (int kc = 0; kc < 8; kc++) {
        const unsigned short* ap = qT + aBase + kc * 32 + aOff;
        v8bf lo = *(const v8bf*)(ap);
        v8bf hi = *(const v8bf*)(ap + 16);
        v16bf a;
#pragma unroll
        for (int i = 0; i < 8; i++) { a[i] = lo[i]; a[i + 8] = hi[i]; }
        afr[kc] = a;
    }

    float rs[8], aR[8], aG[8], aB[8];
#pragma unroll
    for (int j = 0; j < 8; j++) { rs[j] = 0.f; aR[j] = 0.f; aG[j] = 0.f; aB[j] = 0.f; }

    const int bcol = lane & 15;                          // B layout: lane -> key column
    const int bOff = (lane < 16) ? 0 : 8;                // lanes 0-15: K 0..7,16..23, lanes 16-31: K 8..15,24..31

    for (int q0 = wave * 16; q0 < S_; q0 += 64) {
        const int col = q0 + bcol;
        const float2 kv = kq[pcBase + col];              // {betaQ + maskOff, alphaQ}
        const float4 iv = imgPk[bS + col];               // {R, G, B, -}

        // Load ALL B fragments for this tile first: one load clause, one wait,
        // then the WMMAs issue back-to-back.
        const unsigned short* kbase = kT + ((size_t)bS + col) * C_ + bOff;
        v16bf bfr[8];
#pragma unroll
        for (int kc = 0; kc < 8; kc++) {
            v8bf blo = *(const v8bf*)(kbase + kc * 32);
            v8bf bhi = *(const v8bf*)(kbase + kc * 32 + 16);
            v16bf bv;
#pragma unroll
            for (int i = 0; i < 8; i++) { bv[i] = blo[i]; bv[i + 8] = bhi[i]; }
            bfr[kc] = bv;
        }

        if (q0 + 64 < S_)                                // prefetch next key tile row
            __builtin_prefetch(kT + ((size_t)bS + q0 + 64 + bcol) * C_, 0, 1);

        // Two independent accumulator chains -> pipelined WMMA issue.
        v8f acc0 = {}, acc1 = {};
#pragma unroll
        for (int kc = 0; kc < 8; kc += 2) {
            acc0 = __builtin_amdgcn_wmma_f32_16x16x32_bf16(
                false, afr[kc],     false, bfr[kc],     (short)0, acc0, false, false);
            acc1 = __builtin_amdgcn_wmma_f32_16x16x32_bf16(
                false, afr[kc + 1], false, bfr[kc + 1], (short)0, acc1, false, false);
        }

        // branch-free fixed-shift softmax accumulation (co-executes with WMMA)
#pragma unroll
        for (int j = 0; j < 8; j++) {
            float g = acc0[j] + acc1[j];
            float corr = (g + cj[j] - kv.x) * (raP[j] * kv.y);
            float e = __expf(corr - SHIFT);              // masked: underflows to 0
            rs[j] += e;
            aR[j] += e * iv.x;
            aG[j] += e * iv.y;
            aB[j] += e * iv.z;
        }
    }

    // one-time reduction across each 16-lane group; group leaders publish to LDS
#pragma unroll
    for (int j = 0; j < 8; j++) {
        rs[j] = grpsum16(rs[j]);
        aR[j] = grpsum16(aR[j]);
        aG[j] = grpsum16(aG[j]);
        aB[j] = grpsum16(aB[j]);
    }
    if ((lane & 15) == 0) {
#pragma unroll
        for (int j = 0; j < 8; j++) {
            int r = j + hh * 8;
            mS[wave][r] = rs[j];
            mCol[wave][r][0] = aR[j];
            mCol[wave][r][1] = aG[j];
            mCol[wave][r][2] = aB[j];
        }
    }
    __syncthreads();

    // cross-wave merge (shared fixed shift => plain sums); write owner-matched pixels
    if (tid < 16) {
        int r = tid;
        if (owner[bS + p0 + r] == cls) {
            float Ssum = 0.f, cR = 0.f, cG = 0.f, cB = 0.f;
            for (int w2 = 0; w2 < 4; w2++) {
                Ssum += mS[w2][r];
                cR += mCol[w2][r][0];
                cG += mCol[w2][r][1];
                cB += mCol[w2][r][2];
            }
            float inv = 1.f / Ssum;
            out[((size_t)b * 3 + 0) * S_ + p0 + r] = cR * inv;
            out[((size_t)b * 3 + 1) * S_ + p0 + r] = cG * inv;
            out[((size_t)b * 3 + 2) * S_ + p0 + r] = cB * inv;
        }
    }
}

// ---------- launcher ----------
extern "C" void kernel_launch(void* const* d_in, const int* in_sizes, int n_in,
                              void* d_out, int out_size, void* d_ws, size_t ws_size,
                              hipStream_t stream) {
    (void)in_sizes; (void)n_in; (void)out_size; (void)ws_size;
    const float* colorMask = (const float*)d_in[0];   // [B,12,S]
    const float* rf        = (const float*)d_in[1];   // color features [B,C,S] (keys)
    const float* img       = (const float*)d_in[2];   // [B,3,S]
    const float* cf        = (const float*)d_in[3];   // comp features [B,C,S] (queries)
    const float* compMask  = (const float*)d_in[4];   // [B,12,S]
    float* out = (float*)d_out;

    char* w = (char*)d_ws;
    size_t off = 0;
    auto carve = [&](size_t bytes) -> char* {
        off = (off + 255) & ~(size_t)255;
        char* p = w + off;
        off += bytes;
        return p;
    };
    unsigned short* qT = (unsigned short*)carve((size_t)B_ * S_ * C_ * 2);
    unsigned short* kT = (unsigned short*)carve((size_t)B_ * S_ * C_ * 2);
    float*  muC    = (float*)carve((size_t)B_ * NCLS * C_ * 4);
    float*  muL    = (float*)carve((size_t)B_ * NCLS * C_ * 4);
    float*  stats  = (float*)carve((size_t)B_ * NCLS * 8 * 4);
    float*  betaP  = (float*)carve((size_t)B_ * NCLS * S_ * 4);
    float*  alphaP = (float*)carve((size_t)B_ * NCLS * S_ * 4);
    float2* kq     = (float2*)carve((size_t)B_ * NCLS * S_ * 8);
    float4* imgPk  = (float4*)carve((size_t)B_ * S_ * 16);
    int*    owner  = (int*)carve((size_t)B_ * S_ * 4);
    float*  normC2 = (float*)carve((size_t)B_ * S_ * 4);
    float*  normL2 = (float*)carve((size_t)B_ * S_ * 4);

    prep_kernel<<<(B_ * S_) / 256, 256, 0, stream>>>(cf, rf, normC2, normL2, qT, kT);
    stats_kernel<<<dim3(NCLS - 1, B_), 256, 0, stream>>>(cf, rf, compMask, colorMask, muC, muL, stats);
    pixstats_kernel<<<dim3(S_ / 256, NCLS - 1, B_), 256, 0, stream>>>(
        cf, rf, muC, muL, stats, colorMask, normC2, normL2, betaP, alphaP, kq);
    owner_kernel<<<(B_ * S_) / 256, 256, 0, stream>>>(compMask, stats, img, owner, imgPk, out);
    attn_kernel<<<dim3(S_ / 16, NCLS - 1, B_), 128, 0, stream>>>(
        qT, kT, betaP, alphaP, kq, imgPk, stats, owner, out);
}
